// GraphEncoder_3221225472134
// MI455X (gfx1250) — matrix operand order repaired and checked by reading, not compiled, not run on hardware
//
#include <hip/hip_runtime.h>

// Problem constants (from reference)
#define NND 50000
#define EE 800000
#define DD 128
#define D2 256
#define LLAYERS 3
#define GG 64
#define GEN_EPS 1e-7f
#define LN_EPS 1e-5f

typedef __attribute__((ext_vector_type(16))) _Float16 v16h;
typedef __attribute__((ext_vector_type(8)))  _Float16 v8h;
typedef __attribute__((ext_vector_type(8)))  float    v8f;

// ---- monotonic float<->uint mapping so atomicMax(u32) == float max ----
__device__ __forceinline__ unsigned fenc(float f) {
    unsigned b = __float_as_uint(f);
    return (b & 0x80000000u) ? ~b : (b | 0x80000000u);
}
__device__ __forceinline__ float fdec(unsigned k) {
    unsigned b = (k & 0x80000000u) ? (k ^ 0x80000000u) : ~k;
    return __uint_as_float(b);
}
#define NEG_INF_KEY 0x007FFFFFu  // fenc(-inf)

// ---- CDNA5 async global->LDS copy (16B per lane) + wait ----------------
// LDS offset is the low 32 bits of the generic pointer (ISA 10.2: LDS
// aperture address maps to its low 32 bits).
__device__ __forceinline__ void async_copy_b128(void* lds_dst, const void* gsrc) {
    unsigned            ldso = (unsigned)(unsigned long long)lds_dst;
    unsigned long long  ga   = (unsigned long long)gsrc;
    asm volatile("global_load_async_to_lds_b128 %0, %1, off"
                 :: "v"(ldso), "v"(ga) : "memory");
}
__device__ __forceinline__ void async_wait0() {
    asm volatile("s_wait_asynccnt 0x0" ::: "memory");
}

// =====================================================================
// Per-layer init: smax keys = -inf, denom = 0, agg = 0  (vec4 stores)
// =====================================================================
__global__ __launch_bounds__(256) void init_layer_kernel(
    unsigned* __restrict__ smaxK, float* __restrict__ denom, float* __restrict__ agg)
{
    int i = blockIdx.x * 256 + threadIdx.x;          // over NND*DD/4
    if (i >= NND * DD / 4) return;
    uint4 k4 = make_uint4(NEG_INF_KEY, NEG_INF_KEY, NEG_INF_KEY, NEG_INF_KEY);
    float4 z4 = make_float4(0.f, 0.f, 0.f, 0.f);
    ((uint4*)smaxK)[i] = k4;
    ((float4*)denom)[i] = z4;
    ((float4*)agg)[i]   = z4;
}

// =====================================================================
// One-time: convert W1/W2 (3 layers, f32 row-major) into f16 laid out in
// the WMMA B-fragment order used here (contiguous k, same as A): for tile (nt, kb32), lane L holds its 16
// contiguous f16 (element e <-> kin = (L>>4)*16 + e).
//   pos = ((nt*KB32 + kb32)*32 + (kin>>4)*16 + (n&15))*16 + (kin&15)
// =====================================================================
__global__ __launch_bounds__(256) void convert_weights_kernel(
    const float* __restrict__ W1, const float* __restrict__ W2,
    _Float16* __restrict__ W1f, _Float16* __restrict__ W2f)
{
    int i = blockIdx.x * 256 + threadIdx.x;          // over 2*LLAYERS*32768
    if (i >= 2 * LLAYERS * 32768) return;
    int which = (i >= LLAYERS * 32768);
    int j = which ? (i - LLAYERS * 32768) : i;
    int layer = j >> 15;
    int idx   = j & 32767;
    if (!which) {                                    // W1: K=128 x N=256
        int k = idx >> 8, n = idx & 255;
        int kb32 = k >> 5, kin = k & 31;
        int pos = (((n >> 4) * 4 + kb32) * 32 + (kin >> 4) * 16 + (n & 15)) * 16 + (kin & 15);
        W1f[layer * 32768 + pos] = (_Float16)W1[j];
    } else {                                         // W2: K=256 x N=128
        int k = idx >> 7, n = idx & 127;
        int kb32 = k >> 5, kin = k & 31;
        int pos = (((n >> 4) * 8 + kb32) * 32 + (kin >> 4) * 16 + (n & 15)) * 16 + (kin & 15);
        W2f[layer * 32768 + pos] = (_Float16)W2[j];
    }
}

// =====================================================================
// Edge pass 1: segment max of s = t * (relu(x[src]+e)+eps) per (dst,c)
// =====================================================================
__global__ __launch_bounds__(256) void edge_max_kernel(
    const float* __restrict__ x, const float* __restrict__ ef,
    const long long* __restrict__ eidx, const float* __restrict__ t, int layer,
    unsigned* __restrict__ smaxK)
{
    int idx = blockIdx.x * 256 + threadIdx.x;        // over EE*32
    if (idx >= EE * 32) return;
    int e = idx >> 5;
    int c = (idx & 31) << 2;
    int s = (int)eidx[e];
    int d = (int)eidx[EE + e];
    float tv = t[layer];
    __builtin_prefetch(ef + (size_t)e * DD + c + 8192, 0, 1);
    float4 xv = *(const float4*)(x  + (size_t)s * DD + c);
    float4 ev = *(const float4*)(ef + (size_t)e * DD + c);
    float m0 = fmaxf(xv.x + ev.x, 0.f) + GEN_EPS;
    float m1 = fmaxf(xv.y + ev.y, 0.f) + GEN_EPS;
    float m2 = fmaxf(xv.z + ev.z, 0.f) + GEN_EPS;
    float m3 = fmaxf(xv.w + ev.w, 0.f) + GEN_EPS;
    unsigned* p = smaxK + (size_t)d * DD + c;
    atomicMax(p + 0, fenc(tv * m0));
    atomicMax(p + 1, fenc(tv * m1));
    atomicMax(p + 2, fenc(tv * m2));
    atomicMax(p + 3, fenc(tv * m3));
}

// =====================================================================
// Decode max keys -> float; empty segments (still -inf) -> 0
// =====================================================================
__global__ __launch_bounds__(256) void decode_max_kernel(
    const unsigned* __restrict__ smaxK, float* __restrict__ smaxF)
{
    int i = blockIdx.x * 256 + threadIdx.x;          // over NND*DD
    if (i >= NND * DD) return;
    float f = fdec(smaxK[i]);
    if (!(fabsf(f) <= 3.0e38f)) f = 0.f;             // kills +-inf / NaN
    smaxF[i] = f;
}

// =====================================================================
// Edge pass 2: a = exp(s - smax[dst]); denom += a; agg += a*msg
// =====================================================================
__global__ __launch_bounds__(256) void edge_sum_kernel(
    const float* __restrict__ x, const float* __restrict__ ef,
    const long long* __restrict__ eidx, const float* __restrict__ t, int layer,
    const float* __restrict__ smaxF, float* __restrict__ denom, float* __restrict__ agg)
{
    int idx = blockIdx.x * 256 + threadIdx.x;        // over EE*32
    if (idx >= EE * 32) return;
    int e = idx >> 5;
    int c = (idx & 31) << 2;
    int s = (int)eidx[e];
    int d = (int)eidx[EE + e];
    float tv = t[layer];
    __builtin_prefetch(ef + (size_t)e * DD + c + 8192, 0, 1);
    float4 xv = *(const float4*)(x  + (size_t)s * DD + c);
    float4 ev = *(const float4*)(ef + (size_t)e * DD + c);
    const float4 mx = *(const float4*)(smaxF + (size_t)d * DD + c);
    float m0 = fmaxf(xv.x + ev.x, 0.f) + GEN_EPS;
    float m1 = fmaxf(xv.y + ev.y, 0.f) + GEN_EPS;
    float m2 = fmaxf(xv.z + ev.z, 0.f) + GEN_EPS;
    float m3 = fmaxf(xv.w + ev.w, 0.f) + GEN_EPS;
    float a0 = __expf(tv * m0 - mx.x);
    float a1 = __expf(tv * m1 - mx.y);
    float a2 = __expf(tv * m2 - mx.z);
    float a3 = __expf(tv * m3 - mx.w);
    float* pd = denom + (size_t)d * DD + c;
    float* pa = agg   + (size_t)d * DD + c;
    atomicAdd(pd + 0, a0); atomicAdd(pa + 0, a0 * m0);
    atomicAdd(pd + 1, a1); atomicAdd(pa + 1, a1 * m1);
    atomicAdd(pd + 2, a2); atomicAdd(pa + 2, a2 * m2);
    atomicAdd(pd + 3, a3); atomicAdd(pa + 3, a3 * m3);
}

// =====================================================================
// WMMA fragment loads (all b128 LDS accesses)
// =====================================================================
// A (16-bit, 16x32): lane m = lane&15, hf = lane>>4; contiguous order to
// match B fragments: k = kb+hf*16+0..15
__device__ __forceinline__ v16h load_a_frag(const _Float16* __restrict__ rowp,
                                            int kb, int hf)
{
    union { v16h v; v8h h[2]; } u;
    u.h[0] = *(const v8h*)(rowp + kb + hf * 16);
    u.h[1] = *(const v8h*)(rowp + kb + hf * 16 + 8);
    return u.v;
}
// B from fragment-major staged weights: 16 contiguous f16 per lane.
__device__ __forceinline__ v16h load_b_frag(const _Float16* __restrict__ sWf,
                                            int tile, int lane)
{
    const _Float16* p = sWf + tile * 512 + lane * 16;
    union { v16h v; v8h h[2]; } u;
    u.h[0] = *(const v8h*)p;
    u.h[1] = *(const v8h*)(p + 8);
    return u.v;
}

// =====================================================================
// Node update: MessageNorm + MLP(D->2D, LN, ReLU, 2D->D) via WMMA.
// 256 threads = 8 wave32, 128 nodes/block; one wave owns 16 rows.
// LDS: sW (64KB, W1 then W2, f16 fragment-major, async-staged)
//      shA (32KB h f16) | sG (64KB g f16; doubles as f32 aggv scratch)
//      sStat (2KB).
// =====================================================================
#define SMEM_BYTES (65536 + 32768 + 65536 + 2048)

__global__ __launch_bounds__(256) void node_update_kernel(
    const float* __restrict__ xin, const float* __restrict__ agg,
    const float* __restrict__ denom, const float* __restrict__ msca, int layer,
    const _Float16* __restrict__ W1f, const float* __restrict__ b1,
    const float* __restrict__ g1, const float* __restrict__ bb1,
    const _Float16* __restrict__ W2f, const float* __restrict__ b2,
    float* __restrict__ xout)
{
    extern __shared__ char smem[];
    _Float16* sW    = (_Float16*)smem;                          // 32768 f16
    _Float16* shA   = (_Float16*)(smem + 65536);                // 128x128 f16
    _Float16* sG    = (_Float16*)(smem + 65536 + 32768);        // 128x256 f16
    float*    sStat = (float*)(smem + 65536 + 32768 + 65536);   // 512 f32
    float*    aggvS = (float*)sG;                               // 128x128 f32 scratch

    const int tid   = threadIdx.x;
    const int node0 = blockIdx.x * 128;
    const float scale = msca[layer];

    // zero per-node norm accumulators
    sStat[tid] = 0.f;
    sStat[256 + tid] = 0.f;

    // async-stage W1 fragments (f16, pre-permuted) into LDS: 16B per op
    for (int i = tid * 8; i < 32768; i += 256 * 8)
        async_copy_b128(sW + i, W1f + i);

    __syncthreads();   // sStat zeros visible before atomics

    // pass 1: aggv = agg/max(denom,1e-16); accumulate ||aggv||^2, ||x||^2
    for (int i8 = tid; i8 < 2048; i8 += 256) {       // 8 channels / iter
        int nl = i8 >> 4;
        int c0 = (i8 & 15) << 3;
        int n = node0 + nl;
        float av[8]; float sa = 0.f, sx = 0.f;
        if (n < NND) {
            const float* pd = denom + (size_t)n * DD + c0;
            const float* pa = agg   + (size_t)n * DD + c0;
            const float* px = xin   + (size_t)n * DD + c0;
            float4 d0 = *(const float4*)(pd), d1 = *(const float4*)(pd + 4);
            float4 a0 = *(const float4*)(pa), a1 = *(const float4*)(pa + 4);
            float4 x0 = *(const float4*)(px), x1 = *(const float4*)(px + 4);
            const float* dv = (const float*)&d0;     // d0,d1 contiguous on stack
            const float* avp = (const float*)&a0;
            const float* xvp = (const float*)&x0;
            float dd[8] = { d0.x,d0.y,d0.z,d0.w, d1.x,d1.y,d1.z,d1.w };
            float aa[8] = { a0.x,a0.y,a0.z,a0.w, a1.x,a1.y,a1.z,a1.w };
            float xx[8] = { x0.x,x0.y,x0.z,x0.w, x1.x,x1.y,x1.z,x1.w };
            (void)dv; (void)avp; (void)xvp;
#pragma unroll
            for (int e = 0; e < 8; ++e) {
                float v = aa[e] / fmaxf(dd[e], 1e-16f);
                av[e] = v;
                sa += v * v;
                sx += xx[e] * xx[e];
            }
        } else {
#pragma unroll
            for (int e = 0; e < 8; ++e) av[e] = 0.f;
        }
        float* pw = aggvS + nl * 128 + c0;
        *(float4*)(pw)     = make_float4(av[0], av[1], av[2], av[3]);
        *(float4*)(pw + 4) = make_float4(av[4], av[5], av[6], av[7]);
        atomicAdd(&sStat[nl], sa);
        atomicAdd(&sStat[128 + nl], sx);
    }
    __syncthreads();

    // pass 2: h = x + scale * ||x|| * aggv/||aggv||  -> f16 A-tile
    for (int i8 = tid; i8 < 2048; i8 += 256) {
        int nl = i8 >> 4;
        int c0 = (i8 & 15) << 3;
        int n = node0 + nl;
        v8h hv;
        if (n < NND) {
            float coef = scale * sqrtf(sStat[128 + nl])
                       / fmaxf(sqrtf(sStat[nl]), 1e-12f);
            const float* pw = aggvS + nl * 128 + c0;
            const float* px = xin + (size_t)n * DD + c0;
            float4 v0 = *(const float4*)(pw), v1 = *(const float4*)(pw + 4);
            float4 x0 = *(const float4*)(px), x1 = *(const float4*)(px + 4);
            float aa[8] = { v0.x,v0.y,v0.z,v0.w, v1.x,v1.y,v1.z,v1.w };
            float xx[8] = { x0.x,x0.y,x0.z,x0.w, x1.x,x1.y,x1.z,x1.w };
#pragma unroll
            for (int e = 0; e < 8; ++e) hv[e] = (_Float16)(xx[e] + coef * aa[e]);
        } else {
#pragma unroll
            for (int e = 0; e < 8; ++e) hv[e] = (_Float16)0.f;
        }
        *(v8h*)(shA + nl * 128 + c0) = hv;
    }
    async_wait0();     // our W1 async stores into LDS are done
    __syncthreads();   // everyone's are

    const int wave = tid >> 5, lane = tid & 31;
    const int mrow = lane & 15, hf = lane >> 4, ncol = lane & 15;

    // GEMM1: (16x128) x (128x256) per wave; A fragments hoisted over nt
    {
        const _Float16* arow = shA + (wave * 16 + mrow) * 128;
        v16h afr[4];
#pragma unroll
        for (int kk = 0; kk < 4; ++kk) afr[kk] = load_a_frag(arow, kk * 32, hf);
        for (int nt = 0; nt < 16; ++nt) {
            v8f acc = {};
#pragma unroll
            for (int kk = 0; kk < 4; ++kk) {
                v16h b = load_b_frag(sW, nt * 4 + kk, lane);
                acc = __builtin_amdgcn_wmma_f32_16x16x32_f16(
                    false, afr[kk], false, b, (short)0, acc, false, false);
            }
            float bias = b1[nt * 16 + ncol];
#pragma unroll
            for (int r = 0; r < 8; ++r) {
                int mm = r + hf * 8;
                sG[(wave * 16 + mm) * 256 + nt * 16 + ncol] = (_Float16)(acc[r] + bias);
            }
        }
    }
    __syncthreads();   // sG complete; sW no longer read

    // async-stage W2 fragments; meanwhile compute per-node LN stats on sG
    for (int i = tid * 8; i < 32768; i += 256 * 8)
        async_copy_b128(sW + i, W2f + i);
    if (tid < 128) {
        float s = 0.f, q = 0.f;
        for (int c = 0; c < 256; c += 8) {
            v8h v = *(const v8h*)(sG + tid * 256 + c);
#pragma unroll
            for (int e = 0; e < 8; ++e) {
                float f = (float)v[e];
                s += f; q += f * f;
            }
        }
        float mu  = s * (1.f / 256.f);
        float var = q * (1.f / 256.f) - mu * mu;
        sStat[256 + tid] = mu;
        sStat[384 + tid] = rsqrtf(var + LN_EPS);
    }
    async_wait0();
    __syncthreads();

    // LayerNorm + ReLU in place (8 f16 / iter)
    for (int i8 = tid; i8 < 4096; i8 += 256) {
        int nl = i8 >> 5;
        int c0 = (i8 & 31) << 3;
        float mu = sStat[256 + nl], rv = sStat[384 + nl];
        v8h v = *(const v8h*)(sG + nl * 256 + c0);
        float4 gg0 = *(const float4*)(g1 + c0),  gg1 = *(const float4*)(g1 + c0 + 4);
        float4 bbv0 = *(const float4*)(bb1 + c0), bbv1 = *(const float4*)(bb1 + c0 + 4);
        float ga[8] = { gg0.x,gg0.y,gg0.z,gg0.w, gg1.x,gg1.y,gg1.z,gg1.w };
        float ba[8] = { bbv0.x,bbv0.y,bbv0.z,bbv0.w, bbv1.x,bbv1.y,bbv1.z,bbv1.w };
#pragma unroll
        for (int e = 0; e < 8; ++e) {
            float f = (float)v[e];
            f = (f - mu) * rv * ga[e] + ba[e];
            v[e] = (_Float16)fmaxf(f, 0.f);
        }
        *(v8h*)(sG + nl * 256 + c0) = v;
    }
    __syncthreads();

    // GEMM2: (16x256) x (256x128) per wave; 8 A fragments hoisted
    {
        const _Float16* arow = sG + (wave * 16 + mrow) * 256;
        v16h afr[8];
#pragma unroll
        for (int kk = 0; kk < 8; ++kk) afr[kk] = load_a_frag(arow, kk * 32, hf);
        for (int nt = 0; nt < 8; ++nt) {
            v8f acc = {};
#pragma unroll
            for (int kk = 0; kk < 8; ++kk) {
                v16h b = load_b_frag(sW, nt * 8 + kk, lane);
                acc = __builtin_amdgcn_wmma_f32_16x16x32_f16(
                    false, afr[kk], false, b, (short)0, acc, false, false);
            }
            float bias = b2[nt * 16 + ncol];
#pragma unroll
            for (int r = 0; r < 8; ++r) {
                int mm = r + hf * 8;
                int n = node0 + wave * 16 + mm;
                if (n < NND) xout[(size_t)n * DD + nt * 16 + ncol] = acc[r] + bias;
            }
        }
    }
}

// =====================================================================
// Zero pooled / counts
// =====================================================================
__global__ __launch_bounds__(256) void zero_pool_kernel(
    float* __restrict__ pooled, float* __restrict__ counts)
{
    int i = blockIdx.x * 256 + threadIdx.x;
    if (i < GG * DD) pooled[i] = 0.f;
    if (i < GG)      counts[i] = 0.f;
}

// =====================================================================
// Final per-node LayerNorm -> out_local; scatter into pooled sums
// =====================================================================
__global__ __launch_bounds__(128) void local_ln_kernel(
    const float* __restrict__ x, const float* __restrict__ g, const float* __restrict__ b,
    const long long* __restrict__ batch, float* __restrict__ out,
    float* __restrict__ pooled, float* __restrict__ counts)
{
    __shared__ float rs[4], rq[4];
    int n = blockIdx.x, c = threadIdx.x;
    float v = x[(size_t)n * DD + c];
    float s = v, q = v * v;
#pragma unroll
    for (int off = 16; off; off >>= 1) {
        s += __shfl_xor(s, off, 32);
        q += __shfl_xor(q, off, 32);
    }
    int w = c >> 5;
    if ((c & 31) == 0) { rs[w] = s; rq[w] = q; }
    __syncthreads();
    if (c == 0) {
        float S = rs[0] + rs[1] + rs[2] + rs[3];
        float Q = rq[0] + rq[1] + rq[2] + rq[3];
        float mu = S * (1.f / DD);
        rs[0] = mu;
        rq[0] = rsqrtf(Q * (1.f / DD) - mu * mu + LN_EPS);
    }
    __syncthreads();
    float o = (v - rs[0]) * rq[0] * g[c] + b[c];
    out[(size_t)n * DD + c] = o;
    int bg = (int)batch[n];
    atomicAdd(&pooled[bg * DD + c], o);
    if (c == 0) atomicAdd(&counts[bg], 1.f);
}

// =====================================================================
// Pooled head: mean, lin(D,D), LayerNorm. 64x128x128 — negligible work.
// =====================================================================
__global__ __launch_bounds__(128) void pool_head_kernel(
    const float* __restrict__ pooled, const float* __restrict__ counts,
    const float* __restrict__ W, const float* __restrict__ bias,
    const float* __restrict__ g, const float* __restrict__ b,
    float* __restrict__ out)
{
    __shared__ float sp[DD];
    __shared__ float rs[4], rq[4];
    int gid = blockIdx.x, c = threadIdx.x;
    float cnt = fmaxf(counts[gid], 1.f);
    sp[c] = pooled[gid * DD + c] / cnt;
    __syncthreads();
    float acc = bias[c];
    for (int k = 0; k < DD; ++k) acc += sp[k] * W[k * DD + c];
    float s = acc, q = acc * acc;
#pragma unroll
    for (int off = 16; off; off >>= 1) {
        s += __shfl_xor(s, off, 32);
        q += __shfl_xor(q, off, 32);
    }
    int w = c >> 5;
    if ((c & 31) == 0) { rs[w] = s; rq[w] = q; }
    __syncthreads();
    if (c == 0) {
        float S = rs[0] + rs[1] + rs[2] + rs[3];
        float Q = rq[0] + rq[1] + rq[2] + rq[3];
        float mu = S * (1.f / DD);
        rs[0] = mu;
        rq[0] = rsqrtf(Q * (1.f / DD) - mu * mu + LN_EPS);
    }
    __syncthreads();
    out[gid * DD + c] = (acc - rs[0]) * rq[0] * g[c] + b[c];
}

// =====================================================================
// Host launcher
// =====================================================================
extern "C" void kernel_launch(void* const* d_in, const int* in_sizes, int n_in,
                              void* d_out, int out_size, void* d_ws, size_t ws_size,
                              hipStream_t stream)
{
    const float*     x0     = (const float*)d_in[0];
    const float*     ef     = (const float*)d_in[1];
    const float*     t      = (const float*)d_in[2];
    const float*     mscale = (const float*)d_in[3];
    const float*     W1     = (const float*)d_in[4];
    const float*     b1     = (const float*)d_in[5];
    const float*     g1     = (const float*)d_in[6];
    const float*     bb1    = (const float*)d_in[7];
    const float*     W2     = (const float*)d_in[8];
    const float*     b2     = (const float*)d_in[9];
    const float*     linW   = (const float*)d_in[10];
    const float*     linb   = (const float*)d_in[11];
    const float*     n1g    = (const float*)d_in[12];
    const float*     n1b    = (const float*)d_in[13];
    const float*     n2g    = (const float*)d_in[14];
    const float*     n2b    = (const float*)d_in[15];
    const long long* eidx   = (const long long*)d_in[16];  // int64 [2,E]
    const long long* batch  = (const long long*)d_in[17];  // int64 [N]

    float* out_local = (float*)d_out;
    float* out_glob  = out_local + (size_t)NND * DD;

    // workspace carve-up (~128.5 MB; node state stays L2-resident)
    char*  ws  = (char*)d_ws;
    size_t ND4 = (size_t)NND * DD * sizeof(float);
    float*    xA     = (float*)(ws + 0 * ND4);
    float*    xB     = (float*)(ws + 1 * ND4);
    unsigned* smaxK  = (unsigned*)(ws + 2 * ND4);
    float*    smaxF  = (float*)smaxK;              // decoded in place
    float*    denom  = (float*)(ws + 3 * ND4);
    float*    aggp   = (float*)(ws + 4 * ND4);
    size_t    off    = 5 * ND4;
    float*    pooled = (float*)(ws + off);
    float*    counts = pooled + GG * DD;
    off += 65536;
    _Float16* W1f    = (_Float16*)(ws + off); off += (size_t)LLAYERS * 32768 * 2;
    _Float16* W2f    = (_Float16*)(ws + off); off += (size_t)LLAYERS * 32768 * 2;

    const int gridInit = (NND * DD / 4 + 255) / 256;
    const int gridEdge = (EE * 32 + 255) / 256;
    const int gridDec  = (NND * DD + 255) / 256;
    const int gridNode = (NND + 127) / 128;
    const int gridCvt  = (2 * LLAYERS * 32768 + 255) / 256;

    convert_weights_kernel<<<gridCvt, 256, 0, stream>>>(W1, W2, W1f, W2f);

    const float* xin  = x0;
    float*       xout = xA;
    for (int l = 0; l < LLAYERS; ++l) {
        init_layer_kernel<<<gridInit, 256, 0, stream>>>(smaxK, denom, aggp);
        edge_max_kernel<<<gridEdge, 256, 0, stream>>>(xin, ef, eidx, t, l, smaxK);
        decode_max_kernel<<<gridDec, 256, 0, stream>>>(smaxK, smaxF);
        edge_sum_kernel<<<gridEdge, 256, 0, stream>>>(xin, ef, eidx, t, l, smaxF, denom, aggp);
        node_update_kernel<<<gridNode, 256, SMEM_BYTES, stream>>>(
            xin, aggp, denom, mscale, l,
            W1f + (size_t)l * 32768, b1 + (size_t)l * D2,
            g1 + (size_t)l * D2, bb1 + (size_t)l * D2,
            W2f + (size_t)l * 32768, b2 + (size_t)l * DD,
            xout);
        xin  = xout;
        xout = (xout == xA) ? xB : xA;
    }

    zero_pool_kernel<<<(GG * DD + 255) / 256, 256, 0, stream>>>(pooled, counts);
    local_ln_kernel<<<NND, 128, 0, stream>>>(xin, n1g, n1b, batch, out_local, pooled, counts);
    pool_head_kernel<<<GG, 128, 0, stream>>>(pooled, counts, linW, linb, n2g, n2b, out_glob);
}
